// FeedBack_66941360276244
// MI455X (gfx1250) — matrix operand order repaired and checked by reading, not compiled, not run on hardware
//
#include <hip/hip_runtime.h>
#include <math.h>

typedef __bf16 bf16_t;
typedef __attribute__((ext_vector_type(8)))  bf16_t v8bf;
typedef __attribute__((ext_vector_type(16))) bf16_t v16bf;
typedef __attribute__((ext_vector_type(8)))  float  v8f;

#define BB 64
#define TT 128
#define UU 1024
#define G4 4096
#define NSTEPS 16

// ---------------------------------------------------------------------------
// utility: zero a u32 region (graph-capture-safe init for zero-state buffers)
// ---------------------------------------------------------------------------
__global__ void zero_u32(unsigned int* __restrict__ p, int n) {
    int i = blockIdx.x * blockDim.x + threadIdx.x;
    if (i < n) p[i] = 0u;
}

// ---------------------------------------------------------------------------
// weight prep: fp32 [K][4096] -> bf16 transposed [4096][K], LDS 32x32 tiles.
// Rows k < 1024 come from src0, rows k >= 1024 from src1 (fused k2|r2).
// grid = (4096/32, K/32), block = (32,8)
// ---------------------------------------------------------------------------
__global__ __launch_bounds__(256) void transpose_conv(
    const float* __restrict__ src0, const float* __restrict__ src1,
    bf16_t* __restrict__ dst, int K)
{
    __shared__ float tile[32][33];
    const int tn = blockIdx.x * 32;   // n tile (fast dim of source)
    const int tk = blockIdx.y * 32;   // k tile (rows of source)
    const int tx = threadIdx.x;       // 0..31
    const int ty = threadIdx.y;       // 0..7
#pragma unroll
    for (int i = 0; i < 32; i += 8) {
        int k = tk + ty + i;
        const float* s = (k < 1024) ? (src0 + (size_t)k * G4)
                                    : (src1 + (size_t)(k - 1024) * G4);
        tile[ty + i][tx] = s[tn + tx];                 // coalesced along n
    }
    __syncthreads();
#pragma unroll
    for (int i = 0; i < 32; i += 8) {
        int n = tn + ty + i;
        dst[(size_t)n * K + tk + tx] = (bf16_t)tile[tx][ty + i];  // coalesced along k
    }
}

// ---------------------------------------------------------------------------
// One LSTM cell step, fully fused:
//   z = A @ Wt^T + bias (+ x*k1 rank-1 term for layer1), gates, c/h update.
//
// Workgroup = 256 threads = 8 waves. wave = ks*4 + gate:
//   gate  = wave & 3   (i,f,g,o column block)
//   ks    = wave >> 2  (K-slice 0/1) -> halves the dependent WMMA chain
//
// Templated on K so the inner loop trip count is a compile-time constant and
// all slice/pointer selection is hoisted out of the loop (uniform, no exec
// masking): the loop body is pure b128 loads + v_wmma_f32_16x16x32_bf16.
// For K==2048 the slice boundary coincides with the Ax|Ah boundary, so each
// wave reads exactly one A source. grid = (64 u-tiles, 4 batch-tiles).
// ---------------------------------------------------------------------------
template <int K, bool HASX>
__global__ __launch_bounds__(256) void lstm_step(
    const bf16_t* __restrict__ Ah,     // [64][1024] recurrent h_{t-1}
    const bf16_t* __restrict__ Ax,     // [64][1024] layer-1 output (K==2048 only)
    const bf16_t* __restrict__ Wt,     // [4096][K]
    const float*  __restrict__ bias,   // [4096]
    const float*  __restrict__ k1,     // [4096] input kernel (HASX only)
    const float*  __restrict__ xbase,  // layer1 scalar input, x_b = xbase[b*xstride]
    int xstride,
    const float*  __restrict__ cprev,  // [64][1024]
    float*        __restrict__ cout,   // [64][1024]
    bf16_t*       __restrict__ hout)   // [64][1024]
{
    __shared__ float zbuf[8][16][16];

    const int lane = threadIdx.x & 31;
    const int wave = threadIdx.x >> 5;         // 0..7
    const int gate = wave & 3;                 // i,f,g,o
    const int ks   = wave >> 2;                // K slice 0/1
    const int ut   = blockIdx.x;               // unit tile 0..63
    const int rt   = blockIdx.y;               // batch tile 0..3
    const int half = (lane < 16) ? 0 : 1;
    const int l16  = lane & 15;

    constexpr int Khalf = K / 2;
    const int kbeg = ks * Khalf;

    const int arow = rt * 16 + l16;                        // batch row for A frag
    const int ncol = gate * 1024 + ut * 16 + l16;          // output column for B frag

    // Hoisted (uniform-per-wave) A source selection:
    //   K==2048: slice 0 -> Ax rows [0,1024), slice 1 -> Ah rows [0,1024)
    //   K==1024: both slices -> Ah, offset by kbeg
    const bf16_t* Asrc;
    int akoff;
    if (K == 2048) { Asrc = (ks == 0) ? Ax : Ah; akoff = 0; }
    else           { Asrc = Ah;                  akoff = kbeg; }

    // A fragment (16x32 bf16): lane holds row arow, K chunks {kb..+7, kb+16..+23}
    const bf16_t* ap = Asrc + (size_t)arow * 1024 + akoff + half * 8;
    // B fragment (32x16 bf16): lane holds col ncol, K chunks {kb..+7, kb+16..+23}
    const bf16_t* bp = Wt + (size_t)ncol * K + kbeg + half * 8;

    v8f acc = {0.f, 0.f, 0.f, 0.f, 0.f, 0.f, 0.f, 0.f};

#pragma unroll 4
    for (int i = 0; i < Khalf / 32; ++i) {
        v8bf a_lo = *(const v8bf*)(ap);
        v8bf a_hi = *(const v8bf*)(ap + 16);
        v8bf b_lo = *(const v8bf*)(bp);
        v8bf b_hi = *(const v8bf*)(bp + 16);

        v16bf af, bf;
#pragma unroll
        for (int e = 0; e < 8; ++e) {
            af[e] = a_lo[e]; af[e + 8] = a_hi[e];
            bf[e] = b_lo[e]; bf[e + 8] = b_hi[e];
        }
        acc = __builtin_amdgcn_wmma_f32_16x16x32_bf16(
            false, af, false, bf, (short)0, acc, false, false);
        ap += 32;
        bp += 32;
    }

    // C/D layout: VGPR j -> row (j + half*8), col l16
#pragma unroll
    for (int j = 0; j < 8; ++j)
        zbuf[wave][j + half * 8][l16] = acc[j];
    __syncthreads();

    // elementwise gate math: 256 threads, one (b,u) element each
    const int c = threadIdx.x & 15;
    const int r = threadIdx.x >> 4;            // 0..15
    const int b = rt * 16 + r;
    const int u = ut * 16 + c;

    float zi = zbuf[0][r][c] + zbuf[4][r][c] + bias[u];
    float zf = zbuf[1][r][c] + zbuf[5][r][c] + bias[u + 1024];
    float zg = zbuf[2][r][c] + zbuf[6][r][c] + bias[u + 2048];
    float zo = zbuf[3][r][c] + zbuf[7][r][c] + bias[u + 3072];

    if constexpr (HASX) {  // layer1 rank-1 input term: z += x_b * k1[:]
        float xv = xbase[(size_t)b * xstride];
        zi += xv * k1[u];
        zf += xv * k1[u + 1024];
        zg += xv * k1[u + 2048];
        zo += xv * k1[u + 3072];
    }

    float si = 1.f / (1.f + __expf(-zi));
    float sf = 1.f / (1.f + __expf(-zf));
    float so = 1.f / (1.f + __expf(-zo));
    float gg = tanhf(zg);

    float cold = cprev[(size_t)b * UU + u];
    float cnew = sf * cold + si * gg;
    float hnew = so * tanhf(cnew);

    cout[(size_t)b * UU + u] = cnew;
    hout[(size_t)b * UU + u] = (bf16_t)hnew;
}

// ---------------------------------------------------------------------------
// Dense head over the whole sequence: pred[b,t] = h2seq[t][b][:] . wd + bd
// One wave per output; lanes stride the 1024-dot (coalesced), shuffle-reduce.
// outbase points at d_out + it*128 ; flat index b*(16*128) + t
// grid = 1024 blocks x 256 threads = 8192 waves = 64*128 outputs
// ---------------------------------------------------------------------------
__global__ __launch_bounds__(256) void dense_k(
    const bf16_t* __restrict__ h2seq, const float* __restrict__ wd,
    const float* __restrict__ bd, float* __restrict__ outbase)
{
    const int wid  = (blockIdx.x * 256 + threadIdx.x) >> 5;   // 0..8191
    const int lane = threadIdx.x & 31;
    const int b = wid >> 7;
    const int t = wid & 127;
    const bf16_t* hp = h2seq + ((size_t)t * BB + b) * UU;

    float acc = 0.f;
#pragma unroll 4
    for (int u = lane; u < UU; u += 32) acc += (float)hp[u] * wd[u];
#pragma unroll
    for (int off = 16; off > 0; off >>= 1) acc += __shfl_down(acc, off, 32);

    if (lane == 0) outbase[(size_t)b * (NSTEPS * TT) + t] = acc + bd[0];
}

// ---------------------------------------------------------------------------
extern "C" void kernel_launch(void* const* d_in, const int* in_sizes, int n_in,
                              void* d_out, int out_size, void* d_ws, size_t ws_size,
                              hipStream_t stream) {
    const float* inputs = (const float*)d_in[0];   // [64,128,1]
    const float* k1     = (const float*)d_in[1];   // [1,4096]
    const float* r1     = (const float*)d_in[2];   // [1024,4096]
    const float* b1     = (const float*)d_in[3];   // [4096]
    const float* k2     = (const float*)d_in[4];   // [1024,4096]
    const float* r2     = (const float*)d_in[5];   // [1024,4096]
    const float* b2     = (const float*)d_in[6];   // [4096]
    const float* wd     = (const float*)d_in[7];   // [1024,1]
    const float* bd     = (const float*)d_in[8];   // [1]
    float* outp = (float*)d_out;                   // [64, 16*128] row-major

    // ---- workspace carve-up (all 256B aligned) ----
    char* ws = (char*)d_ws;
    size_t off = 0;
    auto take = [&](size_t bytes) {
        char* p = ws + off;
        off += (bytes + 255) & ~(size_t)255;
        return p;
    };
    bf16_t* w1t   = (bf16_t*)take((size_t)G4 * 1024 * 2);        //  8 MB
    bf16_t* w2t   = (bf16_t*)take((size_t)G4 * 2048 * 2);        // 16 MB
    bf16_t* h1seq = (bf16_t*)take((size_t)TT * BB * UU * 2);     // 16 MB
    bf16_t* h2seq = (bf16_t*)take((size_t)TT * BB * UU * 2);     // 16 MB
    float*  c1    = (float*) take((size_t)BB * UU * 4);
    float*  c2    = (float*) take((size_t)BB * UU * 4);
    bf16_t* zh    = (bf16_t*)take((size_t)BB * UU * 2);          // zero h state
    float*  zc    = (float*) take((size_t)BB * UU * 4);          // zero c state

    // zero states (bit pattern 0 == 0.0 for both f32 and bf16)
    {
        int nzh = (BB * UU * 2) / 4;
        int nzc = (BB * UU * 4) / 4;
        zero_u32<<<(nzh + 255) / 256, 256, 0, stream>>>((unsigned int*)zh, nzh);
        zero_u32<<<(nzc + 255) / 256, 256, 0, stream>>>((unsigned int*)zc, nzc);
    }

    // transpose + bf16-convert weights once (LDS-tiled, coalesced both ways)
    transpose_conv<<<dim3(G4 / 32, 1024 / 32), dim3(32, 8), 0, stream>>>(r1, r1, w1t, 1024);
    transpose_conv<<<dim3(G4 / 32, 2048 / 32), dim3(32, 8), 0, stream>>>(k2, r2, w2t, 2048);

    const dim3 grid(64, 4);
    const size_t HB = (size_t)BB * UU;

    for (int it = 0; it < NSTEPS; ++it) {
        // ---- layer 1 over T steps ----
        // init state: zeros (it==0) else layer2's final (h2seq[T-1], c2)
        for (int t = 0; t < TT; ++t) {
            const bf16_t* hprev = (t == 0) ? ((it == 0) ? zh : h2seq + (size_t)(TT - 1) * HB)
                                           : h1seq + (size_t)(t - 1) * HB;
            const float* cprev  = (t == 0) ? ((it == 0) ? zc : c2) : c1;
            const float* xb;
            int xs;
            if (it == 0) { xb = inputs + t;                       xs = TT; }
            else         { xb = outp + (size_t)(it - 1) * TT + t; xs = NSTEPS * TT; }
            lstm_step<1024, true><<<grid, 256, 0, stream>>>(
                hprev, nullptr, w1t, b1, k1, xb, xs, cprev, c1,
                h1seq + (size_t)t * HB);
        }
        // ---- layer 2 over T steps ----
        // init state: zeros (it==0) else layer1's final (h1seq[T-1], c1)
        for (int t = 0; t < TT; ++t) {
            const bf16_t* hprev = (t == 0) ? ((it == 0) ? zh : h1seq + (size_t)(TT - 1) * HB)
                                           : h2seq + (size_t)(t - 1) * HB;
            const float* cprev  = (t == 0) ? ((it == 0) ? zc : c1) : c2;
            lstm_step<2048, false><<<grid, 256, 0, stream>>>(
                hprev, h1seq + (size_t)t * HB, w2t, b2, nullptr, nullptr, 0,
                cprev, c2, h2seq + (size_t)t * HB);
        }
        // ---- dense head for this iteration's 128 predictions ----
        dense_k<<<(BB * TT) / 8, 256, 0, stream>>>(h2seq, wd, bd, outp + (size_t)it * TT);
    }
}
